// GNN_node_90915867722226
// MI455X (gfx1250) — matrix operand order repaired and checked by reading, not compiled, not run on hardware
//
#include <hip/hip_runtime.h>
#include <hip/hip_bf16.h>

// Problem constants (from reference)
#define NN 100000
#define EE 1600000
#define DD 256
#define HH 512
#define LL 5
#define EFEAT 7

// ---- CDNA5 WMMA types ----
typedef __attribute__((ext_vector_type(16))) __bf16 v16bf;
typedef __attribute__((ext_vector_type(8)))  float  v8f;

union Frag {
  v16bf v;
  unsigned short u[16];
  uint4 q[2];
};

__device__ __forceinline__ unsigned short f2bf(float f) {
  unsigned int u = __float_as_uint(f);
  u += 0x7FFFu + ((u >> 16) & 1u);   // round-to-nearest-even
  return (unsigned short)(u >> 16);
}

__device__ __forceinline__ void atomAddF(float* p, float v) {
  __hip_atomic_fetch_add(p, v, __ATOMIC_RELAXED, __HIP_MEMORY_SCOPE_AGENT);
}

// -------------------- utility kernels --------------------
__global__ void k_zero(float* __restrict__ p, int n4) {
  int i = blockIdx.x * blockDim.x + threadIdx.x;
  if (i < n4) ((float4*)p)[i] = make_float4(0.f, 0.f, 0.f, 0.f);
}

__global__ void k_init_h(const int* __restrict__ x, const float* __restrict__ emb,
                         float* __restrict__ h) {
  int i = blockIdx.x * blockDim.x + threadIdx.x;   // over N*D/4
  if (i >= NN * (DD / 4)) return;
  int r  = i >> 6;                                  // 64 float4 per row
  int c4 = i & 63;
  int t  = x[r];
  ((float4*)h)[(size_t)r * 64 + c4] = ((const float4*)emb)[(size_t)t * 64 + c4];
}

// W1b[l][j][k] = bf16(W1[l][k][j])  (H x D panel, row-major by output column)
__global__ void k_prep_w1(const float* __restrict__ W1, unsigned short* __restrict__ W1b) {
  int i = blockIdx.x * blockDim.x + threadIdx.x;
  if (i >= LL * HH * DD) return;
  int l = i / (HH * DD);
  int rem = i - l * (HH * DD);
  int j = rem / DD;     // H
  int k = rem % DD;     // D
  W1b[i] = f2bf(W1[(size_t)l * DD * HH + (size_t)k * HH + j]);
}

// W2b[l][d][j] = bf16(W2[l][j][d])  (D x H panel)
__global__ void k_prep_w2(const float* __restrict__ W2, unsigned short* __restrict__ W2b) {
  int i = blockIdx.x * blockDim.x + threadIdx.x;
  if (i >= LL * DD * HH) return;
  int l = i / (DD * HH);
  int rem = i - l * (DD * HH);
  int d = rem / HH;
  int j = rem % HH;
  W2b[i] = f2bf(W2[(size_t)l * HH * DD + (size_t)j * DD + d]);
}

// -------------------- message passing + segment_sum --------------------
// One wave per edge iteration; lane owns 8 columns. We[l] (7x256) lives in
// registers (56 per lane). agg += relu(h[src] + edge_attr@We + be), atomically.
__global__ void __launch_bounds__(256) k_msgagg(
    const float* __restrict__ h, const int* __restrict__ src,
    const int* __restrict__ dst, const float* __restrict__ eattr,
    const float* __restrict__ Wel, const float* __restrict__ bel,
    float* __restrict__ agg) {
  const int lane = threadIdx.x & 31;
  const int cb = lane * 8;
  float wreg[EFEAT][8], bereg[8];
#pragma unroll
  for (int k = 0; k < EFEAT; ++k)
#pragma unroll
    for (int j = 0; j < 8; ++j) wreg[k][j] = Wel[k * DD + cb + j];
#pragma unroll
  for (int j = 0; j < 8; ++j) bereg[j] = bel[cb + j];

  int w  = (blockIdx.x * blockDim.x + threadIdx.x) >> 5;
  int nw = (gridDim.x * blockDim.x) >> 5;
  for (int e = w; e < EE; e += nw) {
    int s = src[e];
    int d = dst[e];
    float a[EFEAT];
#pragma unroll
    for (int k = 0; k < EFEAT; ++k) a[k] = eattr[(size_t)e * EFEAT + k];
    const float4* hp = (const float4*)(h + (size_t)s * DD + cb);
    float4 h0 = hp[0], h1 = hp[1];
    float hv[8] = {h0.x, h0.y, h0.z, h0.w, h1.x, h1.y, h1.z, h1.w};
    float* ap = agg + (size_t)d * DD + cb;
#pragma unroll
    for (int j = 0; j < 8; ++j) {
      float m = bereg[j];
#pragma unroll
      for (int k = 0; k < EFEAT; ++k) m = fmaf(a[k], wreg[k][j], m);
      m = fmaxf(m + hv[j], 0.0f);
      atomAddF(ap + j, m);
    }
    int e2 = e + nw;                    // prefetch next gather row (L2-resident)
    if (e2 < EE) {
      int s2 = src[e2];
      __builtin_prefetch((const void*)(h + (size_t)s2 * DD + cb), 0, 0);
    }
  }
}

// -------------------- GEMM1: y1 = ((1+eps)h + agg) @ W1 + b1, + BN1 stats ---
// Block = 256 thr (8 waves) = one 16-row M tile; wave w covers columns
// [w*64, w*64+64) via 4 accumulators. K = 256 -> 8 bf16 WMMA steps.
__global__ void __launch_bounds__(256) k_gemm1(
    const float* __restrict__ h, const float* __restrict__ agg,
    const unsigned short* __restrict__ W1b, const float* __restrict__ b1l,
    const float* __restrict__ eps, int lidx,
    float* __restrict__ y1, float* __restrict__ stats1) {
  __shared__ unsigned short ldsA[16 * DD];
  const int t = threadIdx.x;
  const int row0 = blockIdx.x * 16;
  const float ep1 = 1.0f + eps[lidx];
#pragma unroll
  for (int i = 0; i < 16; ++i) {
    size_t gi = (size_t)(row0 + i) * DD + t;
    ldsA[i * DD + t] = f2bf(fmaf(ep1, h[gi], agg[gi]));
  }
  __syncthreads();

  const int lane = t & 31;
  const int colBase = (t >> 5) * 64;
  const int lhalf = lane >> 4;
  const int l16 = lane & 15;
  const v8f vz = {0.f, 0.f, 0.f, 0.f, 0.f, 0.f, 0.f, 0.f};
  v8f acc[4];
#pragma unroll
  for (int s = 0; s < 4; ++s) acc[s] = vz;

#pragma unroll
  for (int kt = 0; kt < 8; ++kt) {
    const int g0 = kt * 32 + lhalf * 16;       // A: K {g0..g0+15}
    Frag A;
    const uint4* pa = (const uint4*)(ldsA + l16 * DD + g0);
    A.q[0] = pa[0];
    A.q[1] = pa[1];
    const int koff = kt * 32 + lhalf * 16;     // B: K {koff..koff+15}
#pragma unroll
    for (int s = 0; s < 4; ++s) {
      const int c = colBase + s * 16 + l16;
      Frag B;
      const uint4* pb = (const uint4*)(W1b + (size_t)c * DD + koff);
      B.q[0] = pb[0];
      B.q[1] = pb[1];
      acc[s] = __builtin_amdgcn_wmma_f32_16x16x32_bf16(
          false, A.v, false, B.v, (short)0, acc[s], false, false);
    }
  }
#pragma unroll
  for (int s = 0; s < 4; ++s) {
    const int c = colBase + s * 16 + l16;
    const float bb = b1l[c];
    float sum = 0.f, sq = 0.f;
#pragma unroll
    for (int i = 0; i < 8; ++i) {
      int r = row0 + i + lhalf * 8;
      float v = acc[s][i] + bb;
      y1[(size_t)r * HH + c] = v;
      sum += v;
      sq += v * v;
    }
    atomAddF(&stats1[c], sum);
    atomAddF(&stats1[HH + c], sq);
  }
}

// -------------------- GEMM2: y2 = relu(bn1(y1)) @ W2 + b2, + BN2 stats -----
// Block = 128 thr (4 waves) = one 16-row M tile over all 256 output cols.
__global__ void __launch_bounds__(128) k_gemm2(
    const float* __restrict__ y1, const float* __restrict__ sc1,
    const unsigned short* __restrict__ W2b, const float* __restrict__ b2l,
    float* __restrict__ y2, float* __restrict__ stats2) {
  __shared__ unsigned short ldsA[16 * HH];
  const int t = threadIdx.x;
  const int row0 = blockIdx.x * 16;
#pragma unroll 8
  for (int i = 0; i < 64; ++i) {
    int e = i * 128 + t;
    int r = e >> 9;
    int c = e & 511;
    float v = y1[(size_t)(row0 + r) * HH + c];
    v = fmaxf(fmaf(v, sc1[c], sc1[HH + c]), 0.0f);   // fused BN1 + ReLU
    ldsA[e] = f2bf(v);
  }
  __syncthreads();

  const int lane = t & 31;
  const int colBase = (t >> 5) * 64;
  const int lhalf = lane >> 4;
  const int l16 = lane & 15;
  const v8f vz = {0.f, 0.f, 0.f, 0.f, 0.f, 0.f, 0.f, 0.f};
  v8f acc[4];
#pragma unroll
  for (int s = 0; s < 4; ++s) acc[s] = vz;

#pragma unroll 4
  for (int kt = 0; kt < 16; ++kt) {
    const int g0 = kt * 32 + lhalf * 16;
    Frag A;
    const uint4* pa = (const uint4*)(ldsA + l16 * HH + g0);
    A.q[0] = pa[0];
    A.q[1] = pa[1];
    const int koff = kt * 32 + lhalf * 16;
#pragma unroll
    for (int s = 0; s < 4; ++s) {
      const int c = colBase + s * 16 + l16;
      Frag B;
      const uint4* pb = (const uint4*)(W2b + (size_t)c * HH + koff);
      B.q[0] = pb[0];
      B.q[1] = pb[1];
      acc[s] = __builtin_amdgcn_wmma_f32_16x16x32_bf16(
          false, A.v, false, B.v, (short)0, acc[s], false, false);
    }
  }
#pragma unroll
  for (int s = 0; s < 4; ++s) {
    const int c = colBase + s * 16 + l16;
    const float bb = b2l[c];
    float sum = 0.f, sq = 0.f;
#pragma unroll
    for (int i = 0; i < 8; ++i) {
      int r = row0 + i + lhalf * 8;
      float v = acc[s][i] + bb;
      y2[(size_t)r * DD + c] = v;
      sum += v;
      sq += v * v;
    }
    atomAddF(&stats2[c], sum);
    atomAddF(&stats2[DD + c], sq);
  }
}

// -------------------- BN finalize: stats -> (scale, shift) ------------------
__global__ void k_bnfinal(const float* __restrict__ stats, const float* __restrict__ g,
                          const float* __restrict__ b, float* __restrict__ sc, int C) {
  int i = blockIdx.x * blockDim.x + threadIdx.x;
  if (i >= C) return;
  const float invN = 1.0f / (float)NN;
  float mean = stats[i] * invN;
  float var = fmaxf(stats[C + i] * invN - mean * mean, 0.0f);
  float scale = g[i] * rsqrtf(var + 1e-5f);
  sc[i] = scale;
  sc[C + i] = fmaf(-mean, scale, b[i]);
}

// -------------------- apply BN2 (+ReLU) -> h ------------------
__global__ void k_apply(const float* __restrict__ y2, const float* __restrict__ sc2,
                        float* __restrict__ hout, int doRelu) {
  int i = blockIdx.x * blockDim.x + threadIdx.x;   // over N*D/4
  if (i >= NN * (DD / 4)) return;
  int c4 = (i & 63) * 4;
  float4 v = ((const float4*)y2)[i];
  float4 o;
  o.x = fmaf(v.x, sc2[c4 + 0], sc2[DD + c4 + 0]);
  o.y = fmaf(v.y, sc2[c4 + 1], sc2[DD + c4 + 1]);
  o.z = fmaf(v.z, sc2[c4 + 2], sc2[DD + c4 + 2]);
  o.w = fmaf(v.w, sc2[c4 + 3], sc2[DD + c4 + 3]);
  if (doRelu) {
    o.x = fmaxf(o.x, 0.f); o.y = fmaxf(o.y, 0.f);
    o.z = fmaxf(o.z, 0.f); o.w = fmaxf(o.w, 0.f);
  }
  ((float4*)hout)[i] = o;
}

// -------------------- orchestration --------------------
extern "C" void kernel_launch(void* const* d_in, const int* in_sizes, int n_in,
                              void* d_out, int out_size, void* d_ws, size_t ws_size,
                              hipStream_t stream) {
  (void)in_sizes; (void)n_in; (void)out_size; (void)ws_size;
  const int*   x        = (const int*)d_in[0];
  const int*   ei       = (const int*)d_in[1];
  const float* eattr    = (const float*)d_in[2];
  const float* node_emb = (const float*)d_in[4];
  const float* We       = (const float*)d_in[5];
  const float* be       = (const float*)d_in[6];
  const float* W1       = (const float*)d_in[7];
  const float* b1       = (const float*)d_in[8];
  const float* g1       = (const float*)d_in[9];
  const float* bt1      = (const float*)d_in[10];
  const float* W2       = (const float*)d_in[11];
  const float* b2       = (const float*)d_in[12];
  const float* eps      = (const float*)d_in[13];
  const float* g_bn     = (const float*)d_in[14];
  const float* b_bn     = (const float*)d_in[15];
  float* h = (float*)d_out;                       // h lives in d_out (N x D)

  char* ws = (char*)d_ws;
  size_t off = 0;
  auto alloc = [&](size_t bytes) -> void* {
    void* p = ws + off;
    off = (off + bytes + 255) & ~(size_t)255;
    return p;
  };
  float*          agg    = (float*)alloc(sizeof(float) * (size_t)NN * DD);  // also y2
  float*          y1     = (float*)alloc(sizeof(float) * (size_t)NN * HH);
  unsigned short* W1b    = (unsigned short*)alloc(2ull * LL * HH * DD);
  unsigned short* W2b    = (unsigned short*)alloc(2ull * LL * DD * HH);
  float*          stats1 = (float*)alloc(sizeof(float) * 2 * HH);
  float*          stats2 = (float*)alloc(sizeof(float) * 2 * DD);
  float*          sc1    = (float*)alloc(sizeof(float) * 2 * HH);
  float*          sc2    = (float*)alloc(sizeof(float) * 2 * DD);

  const int* src = ei;
  const int* dst = ei + EE;

  k_prep_w1<<<(LL * HH * DD + 255) / 256, 256, 0, stream>>>(W1, W1b);
  k_prep_w2<<<(LL * DD * HH + 255) / 256, 256, 0, stream>>>(W2, W2b);
  k_init_h<<<(NN * (DD / 4) + 255) / 256, 256, 0, stream>>>(x, node_emb, h);

  for (int l = 0; l < LL; ++l) {
    k_zero<<<(NN * (DD / 4) + 255) / 256, 256, 0, stream>>>(agg, NN * (DD / 4));
    k_zero<<<1, 256, 0, stream>>>(stats1, 2 * HH / 4);
    k_zero<<<1, 256, 0, stream>>>(stats2, 2 * DD / 4);

    k_msgagg<<<1024, 256, 0, stream>>>(h, src, dst, eattr,
                                       We + (size_t)l * EFEAT * DD,
                                       be + (size_t)l * DD, agg);

    k_gemm1<<<NN / 16, 256, 0, stream>>>(h, agg, W1b + (size_t)l * HH * DD,
                                         b1 + (size_t)l * HH, eps, l, y1, stats1);
    k_bnfinal<<<(HH + 255) / 256, 256, 0, stream>>>(stats1, g1 + (size_t)l * HH,
                                                    bt1 + (size_t)l * HH, sc1, HH);

    k_gemm2<<<NN / 16, 128, 0, stream>>>(y1, sc1, W2b + (size_t)l * DD * HH,
                                         b2 + (size_t)l * DD, agg /*y2*/, stats2);
    k_bnfinal<<<1, 256, 0, stream>>>(stats2, g_bn + (size_t)l * DD,
                                     b_bn + (size_t)l * DD, sc2, DD);

    k_apply<<<(NN * (DD / 4) + 255) / 256, 256, 0, stream>>>(agg, sc2, h,
                                                             (l != LL - 1) ? 1 : 0);
  }
}
